// TransformerBlock_59090160058510
// MI455X (gfx1250) — compile-verified, matrix operand order checked by reading, not run on hardware
//
#include <hip/hip_runtime.h>

#define D_MODEL 1024
#define NUM_HEADS 16
#define D_KH 64
#define D_FF 4096
#define SEQ 2048
#define BATCH 2
#define MTOK (BATCH * SEQ) /* 4096 */

typedef __attribute__((ext_vector_type(16))) __bf16 v16bf;
typedef __attribute__((ext_vector_type(8)))  float  v8f;
typedef __attribute__((ext_vector_type(4)))  unsigned int u32x4;

union FragBF { v16bf v; u32x4 u[2]; };

__device__ __forceinline__ __bf16 f2bf(float f) {
  union { float f; unsigned u; } a; a.f = f;
  unsigned r = (a.u + 0x7fffu + ((a.u >> 16) & 1u)) >> 16;
  union { unsigned short s; __bf16 b; } o; o.s = (unsigned short)r;
  return o.b;
}

__device__ __forceinline__ v8f wmma_bf16(v16bf a, v16bf b, v8f c) {
  // D = A(16x32 bf16) * B(32x16 bf16) + C(16x16 f32)
  return __builtin_amdgcn_wmma_f32_16x16x32_bf16(false, a, false, b, (short)0, c,
                                                 false, false);
}

// CDNA5 async global->LDS copy (ASYNCcnt-tracked), no VGPR staging.
// LDS byte address = low 32 bits of the flat shared-aperture address.
__device__ __forceinline__ void async_ld_b128(void* lds, const void* gaddr) {
  unsigned l = (unsigned)(size_t)lds;
  asm volatile("global_load_async_to_lds_b128 %0, %1, off"
               :: "v"(l), "v"(gaddr)
               : "memory");
}
__device__ __forceinline__ void wait_async0() {
  asm volatile("s_wait_asynccnt 0x0" ::: "memory");
}

// ---------------------------------------------------------------------------
// Generic bf16 WMMA GEMM: out = A[M,K] * Bt[N,K]^T + bias, with epilogues.
// Block = 256 threads (8 waves), tile 128x128, K-step 64, double-buffered LDS
// filled by GLOBAL_LOAD_ASYNC_TO_LDS_B128 (prefetch tile i+1 while WMMAs
// consume tile i).
// EPI: 0 = bf16 row-major (opt. GELU), 1 = f32 + residual,
//      2 = bf16 [B,H,S,64] head layout (Q,K), 3 = bf16 [B,H,64,S] (V^T).
// ---------------------------------------------------------------------------
template <int EPI, bool GELU>
__global__ __launch_bounds__(256) void gemm_bf16_wmma(
    const __bf16* __restrict__ A, const __bf16* __restrict__ Bt,
    const float* __restrict__ bias, const float* __restrict__ resid,
    void* __restrict__ outp, int N, int K) {
  __shared__ __bf16 As[2][128 * 64];
  __shared__ __bf16 Bs[2][128 * 64];
  const int tid  = threadIdx.x;
  const int lane = tid & 31, wave = tid >> 5;
  const int ln = lane & 15, kg = lane >> 4;
  const int m0 = blockIdx.y * 128, n0 = blockIdx.x * 128;
  const int wm = (wave >> 1) * 32, wn = (wave & 1) * 64;

  // This thread's 4 chunk coordinates (16B chunks, 128x64 bf16 tile).
  v8f acc[2][4] = {};
  const int nk = K >> 6;

  // Prologue: async-fill buffer 0 with K-tile 0.
#pragma unroll
  for (int j = 0; j < 4; ++j) {
    int c = tid + 256 * j;
    int row = c >> 3, col = (c & 7) << 3;
    async_ld_b128(&As[0][row * 64 + col], &A[(size_t)(m0 + row) * K + col]);
    async_ld_b128(&Bs[0][row * 64 + col], &Bt[(size_t)(n0 + row) * K + col]);
  }
  wait_async0();
  __syncthreads();

  for (int kt = 0; kt < nk; ++kt) {
    const int p = kt & 1;
    // Prefetch next K-tile into the other buffer (lands after the barrier
    // that already ordered everyone past its last reader).
    if (kt + 1 < nk) {
      const int k1 = (kt + 1) << 6;
#pragma unroll
      for (int j = 0; j < 4; ++j) {
        int c = tid + 256 * j;
        int row = c >> 3, col = (c & 7) << 3;
        async_ld_b128(&As[p ^ 1][row * 64 + col],
                      &A[(size_t)(m0 + row) * K + k1 + col]);
        async_ld_b128(&Bs[p ^ 1][row * 64 + col],
                      &Bt[(size_t)(n0 + row) * K + k1 + col]);
      }
    }
    // Compute on current buffer.
    const __bf16* Ap = As[p];
    const __bf16* Bp = Bs[p];
#pragma unroll
    for (int ks = 0; ks < 64; ks += 32) {
      FragBF af[2], bfr[4];
#pragma unroll
      for (int mt = 0; mt < 2; ++mt) {
        int r = wm + 16 * mt + ln;
        af[mt].u[0] = *(const u32x4*)&Ap[r * 64 + ks + kg * 8];
        af[mt].u[1] = *(const u32x4*)&Ap[r * 64 + ks + 16 + kg * 8];
      }
#pragma unroll
      for (int nt = 0; nt < 4; ++nt) {
        int r = wn + 16 * nt + ln;
        bfr[nt].u[0] = *(const u32x4*)&Bp[r * 64 + ks + kg * 8];
        bfr[nt].u[1] = *(const u32x4*)&Bp[r * 64 + ks + 16 + kg * 8];
      }
#pragma unroll
      for (int mt = 0; mt < 2; ++mt)
#pragma unroll
        for (int nt = 0; nt < 4; ++nt)
          acc[mt][nt] = wmma_bf16(af[mt].v, bfr[nt].v, acc[mt][nt]);
    }
    wait_async0();    // my prefetch of tile kt+1 has landed
    __syncthreads();  // everyone's prefetch landed; readers of buf p done
  }

#pragma unroll
  for (int mt = 0; mt < 2; ++mt)
#pragma unroll
    for (int nt = 0; nt < 4; ++nt)
#pragma unroll
      for (int r = 0; r < 8; ++r) {
        int m = m0 + wm + 16 * mt + r + 8 * kg;
        int n = n0 + wn + 16 * nt + ln;
        float v = acc[mt][nt][r] + bias[n];
        if (GELU) v = 0.5f * v * (1.0f + erff(v * 0.70710678118f));
        if constexpr (EPI == 0) {
          ((__bf16*)outp)[(size_t)m * N + n] = f2bf(v);
        } else if constexpr (EPI == 1) {
          v += resid[(size_t)m * N + n];
          ((float*)outp)[(size_t)m * N + n] = v;
        } else if constexpr (EPI == 2) {
          int b = m / SEQ, s = m % SEQ, h = n >> 6, dk = n & 63;
          ((__bf16*)outp)[(((size_t)(b * NUM_HEADS + h) * SEQ) + s) * 64 + dk] =
              f2bf(v);
        } else {
          int b = m / SEQ, s = m % SEQ, h = n >> 6, dk = n & 63;
          ((__bf16*)outp)[(((size_t)(b * NUM_HEADS + h) * 64) + dk) * SEQ + s] =
              f2bf(v);
        }
      }
}

// ---------------------------------------------------------------------------
// Flash attention. Block = 128 threads (4 waves); each wave owns 32 q-rows.
// K/V tiles stream into LDS via async copies.
// Q,K: [B*H][S][64] bf16.  Vt: [B*H][64][S] bf16.  ctx: [B*S][1024] bf16.
// ---------------------------------------------------------------------------
__global__ __launch_bounds__(128) void flash_attn_wmma(
    const __bf16* __restrict__ Q, const __bf16* __restrict__ Kb,
    const __bf16* __restrict__ Vt, __bf16* __restrict__ ctx) {
  __shared__ __bf16 Ks[64 * 64];       // [kv][dk]
  __shared__ __bf16 Vs[64 * 64];       // [dk][kv]  (V^T)
  __shared__ __bf16 Pst[4 * 32 * 64];  // per-wave P staging
  const int tid = threadIdx.x;
  const int lane = tid & 31, wave = tid >> 5;
  const int ln = lane & 15, kg = lane >> 4;
  const int bh = blockIdx.y;
  const int q0 = blockIdx.x * 128 + wave * 32;
  const float scale = 0.125f;  // 1/sqrt(64)
  __bf16* Pw = &Pst[wave * 32 * 64];

  const __bf16* Qg = Q + (size_t)bh * SEQ * 64;
  const __bf16* Kg = Kb + (size_t)bh * SEQ * 64;
  const __bf16* Vg = Vt + (size_t)bh * 64 * SEQ;

  FragBF qf[2][2];
#pragma unroll
  for (int mt = 0; mt < 2; ++mt)
#pragma unroll
    for (int ks = 0; ks < 2; ++ks) {
      int row = q0 + 16 * mt + ln;
      int base = ks * 32 + kg * 8;
      qf[mt][ks].u[0] = *(const u32x4*)&Qg[(size_t)row * 64 + base];
      qf[mt][ks].u[1] = *(const u32x4*)&Qg[(size_t)row * 64 + base + 16];
    }

  float mstat[2][8], lstat[2][8];
  v8f o[2][4] = {};
#pragma unroll
  for (int mt = 0; mt < 2; ++mt)
#pragma unroll
    for (int r = 0; r < 8; ++r) {
      mstat[mt][r] = -3.4e38f;
      lstat[mt][r] = 0.0f;
    }

  for (int j = 0; j < SEQ / 64; ++j) {
    __syncthreads();  // previous iteration's readers done with Ks/Vs
#pragma unroll
    for (int c4 = 0; c4 < 4; ++c4) {
      int c = tid + 128 * c4;
      int row = c >> 3, col = (c & 7) << 3;
      async_ld_b128(&Ks[row * 64 + col],
                    &Kg[((size_t)j * 64 + row) * 64 + col]);
      async_ld_b128(&Vs[row * 64 + col],
                    &Vg[(size_t)row * SEQ + (size_t)j * 64 + col]);
    }
    wait_async0();
    __syncthreads();

    // S = Q * K^T  (inner = dk = 64)
    v8f sacc[2][4] = {};
#pragma unroll
    for (int ks = 0; ks < 2; ++ks) {
      FragBF bfr[4];
      int base = ks * 32 + kg * 8;
#pragma unroll
      for (int nt = 0; nt < 4; ++nt) {
        int r = 16 * nt + ln;
        bfr[nt].u[0] = *(const u32x4*)&Ks[r * 64 + base];
        bfr[nt].u[1] = *(const u32x4*)&Ks[r * 64 + base + 16];
      }
#pragma unroll
      for (int mt = 0; mt < 2; ++mt)
#pragma unroll
        for (int nt = 0; nt < 4; ++nt)
          sacc[mt][nt] = wmma_bf16(qf[mt][ks].v, bfr[nt].v, sacc[mt][nt]);
    }

    // online softmax: row r of tile mt lives in one 16-lane half, vgpr r
#pragma unroll
    for (int mt = 0; mt < 2; ++mt)
#pragma unroll
      for (int r = 0; r < 8; ++r) {
        float rowm = sacc[mt][0][r];
#pragma unroll
        for (int nt = 1; nt < 4; ++nt) rowm = fmaxf(rowm, sacc[mt][nt][r]);
        rowm *= scale;
        for (int off = 1; off < 16; off <<= 1)
          rowm = fmaxf(rowm, __shfl_xor(rowm, off, 32));
        float mold = mstat[mt][r];
        float mnew = fmaxf(mold, rowm);
        float alpha = __expf(mold - mnew);
        float rs = 0.0f;
#pragma unroll
        for (int nt = 0; nt < 4; ++nt) {
          float p = __expf(scale * sacc[mt][nt][r] - mnew);
          rs += p;
          Pw[(size_t)(16 * mt + r + 8 * kg) * 64 + 16 * nt + ln] = f2bf(p);
        }
        for (int off = 1; off < 16; off <<= 1) rs += __shfl_xor(rs, off, 32);
        lstat[mt][r] = lstat[mt][r] * alpha + rs;
        mstat[mt][r] = mnew;
#pragma unroll
        for (int nt = 0; nt < 4; ++nt) o[mt][nt][r] *= alpha;
      }
    __syncthreads();  // P visible (also uniform across waves)

    // O += P(32x64) * V(64x64); B operand read from Vs = V^T
#pragma unroll
    for (int ks = 0; ks < 2; ++ks) {
      FragBF pa[2], vb[4];
      int base = ks * 32 + kg * 8;
#pragma unroll
      for (int mt = 0; mt < 2; ++mt) {
        int row = 16 * mt + ln;
        pa[mt].u[0] = *(const u32x4*)&Pw[row * 64 + base];
        pa[mt].u[1] = *(const u32x4*)&Pw[row * 64 + base + 16];
      }
#pragma unroll
      for (int nt = 0; nt < 4; ++nt) {
        int r = 16 * nt + ln;
        vb[nt].u[0] = *(const u32x4*)&Vs[r * 64 + base];
        vb[nt].u[1] = *(const u32x4*)&Vs[r * 64 + base + 16];
      }
#pragma unroll
      for (int mt = 0; mt < 2; ++mt)
#pragma unroll
        for (int nt = 0; nt < 4; ++nt)
          o[mt][nt] = wmma_bf16(pa[mt].v, vb[nt].v, o[mt][nt]);
    }
  }

  const int b = bh / NUM_HEADS, h = bh % NUM_HEADS;
#pragma unroll
  for (int mt = 0; mt < 2; ++mt)
#pragma unroll
    for (int nt = 0; nt < 4; ++nt)
#pragma unroll
      for (int r = 0; r < 8; ++r) {
        int ml = 16 * mt + r + 8 * kg;
        int n = 16 * nt + ln;
        float val = o[mt][nt][r] / lstat[mt][r];
        ctx[((size_t)b * SEQ + q0 + ml) * D_MODEL + h * 64 + n] = f2bf(val);
      }
}

// ---------------------------------------------------------------------------
__global__ __launch_bounds__(256) void layernorm_k(
    const float* __restrict__ in, const float* __restrict__ gamma,
    const float* __restrict__ beta, float* __restrict__ outf,
    __bf16* __restrict__ outb) {
  __shared__ float rs[256], rq[256];
  const int row = blockIdx.x, tid = threadIdx.x;
  const float* x = in + (size_t)row * D_MODEL;
  float s = 0.0f, q = 0.0f;
#pragma unroll
  for (int i = tid; i < D_MODEL; i += 256) {
    float v = x[i];
    s += v;
    q += v * v;
  }
  rs[tid] = s;
  rq[tid] = q;
  __syncthreads();
  for (int st = 128; st > 0; st >>= 1) {
    if (tid < st) {
      rs[tid] += rs[tid + st];
      rq[tid] += rq[tid + st];
    }
    __syncthreads();
  }
  const float mean = rs[0] * (1.0f / D_MODEL);
  const float var = rq[0] * (1.0f / D_MODEL) - mean * mean;
  const float rstd = rsqrtf(var + 1e-5f);
#pragma unroll
  for (int i = tid; i < D_MODEL; i += 256) {
    float y = (x[i] - mean) * rstd * gamma[i] + beta[i];
    outf[(size_t)row * D_MODEL + i] = y;
    if (outb) outb[(size_t)row * D_MODEL + i] = f2bf(y);
  }
}

// W[K,N] f32 -> Wt[N,K] bf16
__global__ __launch_bounds__(256) void transpose_f32_bf16(
    const float* __restrict__ W, __bf16* __restrict__ Wt, int K, int N) {
  __shared__ float tile[32][33];
  const int n0 = blockIdx.x * 32, k0 = blockIdx.y * 32;
  const int tx = threadIdx.x, ty = threadIdx.y;  // (32,8)
#pragma unroll
  for (int j = 0; j < 4; ++j)
    tile[ty + 8 * j][tx] = W[(size_t)(k0 + ty + 8 * j) * N + n0 + tx];
  __syncthreads();
#pragma unroll
  for (int j = 0; j < 4; ++j)
    Wt[(size_t)(n0 + ty + 8 * j) * K + k0 + tx] = f2bf(tile[tx][ty + 8 * j]);
}

__global__ __launch_bounds__(256) void f32_to_bf16_k(const float* __restrict__ x,
                                                     __bf16* __restrict__ y,
                                                     int n) {
  int i = blockIdx.x * blockDim.x + threadIdx.x;
  if (i < n) y[i] = f2bf(x[i]);
}

// ---------------------------------------------------------------------------
extern "C" void kernel_launch(void* const* d_in, const int* in_sizes, int n_in,
                              void* d_out, int out_size, void* d_ws,
                              size_t ws_size, hipStream_t stream) {
  (void)in_sizes; (void)n_in; (void)out_size; (void)ws_size;
  const float* x   = (const float*)d_in[0];
  const float* Wq  = (const float*)d_in[1];
  const float* bq  = (const float*)d_in[2];
  const float* Wk  = (const float*)d_in[3];
  const float* bk  = (const float*)d_in[4];
  const float* Wv  = (const float*)d_in[5];
  const float* bv  = (const float*)d_in[6];
  const float* Wo  = (const float*)d_in[7];
  const float* bo  = (const float*)d_in[8];
  const float* W1  = (const float*)d_in[9];
  const float* b1  = (const float*)d_in[10];
  const float* W2  = (const float*)d_in[11];
  const float* b2  = (const float*)d_in[12];
  const float* g1  = (const float*)d_in[13];
  const float* be1 = (const float*)d_in[14];
  const float* g2  = (const float*)d_in[15];
  const float* be2 = (const float*)d_in[16];

  char* ws = (char*)d_ws;
  const size_t MB = (size_t)1 << 20;
  __bf16* WqT = (__bf16*)(ws + 0 * MB);    // 2MB each
  __bf16* WkT = (__bf16*)(ws + 2 * MB);
  __bf16* WvT = (__bf16*)(ws + 4 * MB);
  __bf16* WoT = (__bf16*)(ws + 6 * MB);
  __bf16* W1T = (__bf16*)(ws + 8 * MB);    // 8MB
  __bf16* W2T = (__bf16*)(ws + 16 * MB);   // 8MB
  __bf16* xbf = (__bf16*)(ws + 24 * MB);   // 8MB
  __bf16* Qb  = (__bf16*)(ws + 32 * MB);   // 8MB
  __bf16* Kb  = (__bf16*)(ws + 40 * MB);   // 8MB
  __bf16* Vtb = (__bf16*)(ws + 48 * MB);   // 8MB
  __bf16* ctx = (__bf16*)(ws + 56 * MB);   // 8MB
  float*  res = (float*)(ws + 64 * MB);    // 16MB (reused as res2)
  float*  x1f = (float*)(ws + 80 * MB);    // 16MB
  __bf16* x1b = (__bf16*)(ws + 96 * MB);   // 8MB
  __bf16* ffh = (__bf16*)(ws + 104 * MB);  // 32MB  (total 136MB)

  dim3 tb(32, 8);
  transpose_f32_bf16<<<dim3(32, 32), tb, 0, stream>>>(Wq, WqT, 1024, 1024);
  transpose_f32_bf16<<<dim3(32, 32), tb, 0, stream>>>(Wk, WkT, 1024, 1024);
  transpose_f32_bf16<<<dim3(32, 32), tb, 0, stream>>>(Wv, WvT, 1024, 1024);
  transpose_f32_bf16<<<dim3(32, 32), tb, 0, stream>>>(Wo, WoT, 1024, 1024);
  transpose_f32_bf16<<<dim3(128, 32), tb, 0, stream>>>(W1, W1T, 1024, 4096);
  transpose_f32_bf16<<<dim3(32, 128), tb, 0, stream>>>(W2, W2T, 4096, 1024);
  f32_to_bf16_k<<<(MTOK * D_MODEL) / 256, 256, 0, stream>>>(x, xbf,
                                                            MTOK * D_MODEL);

  // QKV projections (N=1024), head-layout epilogues
  gemm_bf16_wmma<2, false><<<dim3(8, 32), 256, 0, stream>>>(
      xbf, WqT, bq, nullptr, Qb, 1024, 1024);
  gemm_bf16_wmma<2, false><<<dim3(8, 32), 256, 0, stream>>>(
      xbf, WkT, bk, nullptr, Kb, 1024, 1024);
  gemm_bf16_wmma<3, false><<<dim3(8, 32), 256, 0, stream>>>(
      xbf, WvT, bv, nullptr, Vtb, 1024, 1024);

  flash_attn_wmma<<<dim3(SEQ / 128, BATCH * NUM_HEADS), 128, 0, stream>>>(
      Qb, Kb, Vtb, ctx);

  // attn_out = ctx @ Wo + bo + x  -> res (f32)
  gemm_bf16_wmma<1, false><<<dim3(8, 32), 256, 0, stream>>>(
      ctx, WoT, bo, x, res, 1024, 1024);
  layernorm_k<<<MTOK, 256, 0, stream>>>(res, g1, be1, x1f, x1b);

  // FFN
  gemm_bf16_wmma<0, true><<<dim3(32, 32), 256, 0, stream>>>(
      x1b, W1T, b1, nullptr, ffh, 4096, 1024);
  gemm_bf16_wmma<1, false><<<dim3(8, 32), 256, 0, stream>>>(
      ffh, W2T, b2, x1f, res, 1024, 4096);
  layernorm_k<<<MTOK, 256, 0, stream>>>(res, g2, be2, (float*)d_out, nullptr);
}
